// TactileSGNet_35828617183849
// MI455X (gfx1250) — matrix operand order repaired and checked by reading, not compiled, not run on hardware
//
#include <hip/hip_runtime.h>

#define T_STEPS 2000
#define NN 39
#define CIN 2
#define F1 64
#define NF (NN*F1)          // 2496
#define H1 128
#define H2 256
#define H3 36
#define H3P 48
#define N_EDGES 156
#define DECAYC 0.2f
#define THRESHC 0.5f

typedef __attribute__((ext_vector_type(16))) _Float16 v16h;
typedef __attribute__((ext_vector_type(8)))  float    v8f;
typedef __attribute__((ext_vector_type(4)))  unsigned int v4u;

// ---------- Stage 0: normalized adjacency  A_hat = D^-1/2 A D^-1/2 ----------
__global__ void build_ahat(const int* __restrict__ ei, float* __restrict__ Ahat) {
    __shared__ float sA[NN*NN];
    __shared__ float sdinv[NN];
    int tid = threadIdx.x;
    for (int i = tid; i < NN*NN; i += blockDim.x) sA[i] = 0.f;
    __syncthreads();
    for (int e = tid; e < N_EDGES; e += blockDim.x) {
        int r = ei[e], c = ei[N_EDGES + e];      // A[col,row] += 1
        atomicAdd(&sA[c*NN + r], 1.0f);
    }
    __syncthreads();
    if (tid < NN) {
        float d = 0.f;
        for (int m = 0; m < NN; ++m) d += sA[tid*NN + m];
        sdinv[tid] = (d > 0.f) ? rsqrtf(d) : 0.f;
    }
    __syncthreads();
    for (int i = tid; i < NN*NN; i += blockDim.x) {
        int n = i / NN, m = i % NN;
        Ahat[i] = sdinv[n] * sA[i] * sdinv[m];
    }
}

// ---------- tiny 39x39 matmul: C = A @ B ----------
__global__ void matmul39(const float* __restrict__ A, const float* __restrict__ B,
                         float* __restrict__ C) {
    int id = blockIdx.x*blockDim.x + threadIdx.x;
    if (id >= NN*NN) return;
    int n = id / NN, m = id % NN;
    float s = 0.f;
    for (int p = 0; p < NN; ++p) s += A[n*NN + p] * B[p*NN + m];
    C[id] = s;
}

// ---------- Stage 1: Ycat[n][c8][t], c8 = 2k+c, Y_k = A^k @ x ----------
__global__ void build_ycat(const float* __restrict__ x, const float* __restrict__ A1,
                           const float* __restrict__ A2m, const float* __restrict__ A3m,
                           float* __restrict__ Y) {
    int id = blockIdx.x*blockDim.x + threadIdx.x;
    if (id >= NN*8*T_STEPS) return;
    int t  = id % T_STEPS;
    int r  = id / T_STEPS;
    int c8 = r % 8;
    int n  = r / 8;
    int k = c8 >> 1, c = c8 & 1;
    float v;
    if (k == 0) {
        v = x[(size_t)(n*CIN + c)*T_STEPS + t];
    } else {
        const float* Ak = (k == 1) ? A1 : (k == 2) ? A2m : A3m;
        float s = 0.f;
        for (int m = 0; m < NN; ++m)
            s += Ak[n*NN + m] * x[(size_t)(m*CIN + c)*T_STEPS + t];
        v = s;
    }
    Y[id] = v;
}

// ---------- weight transpose + f16 convert: dst[j][i] = src[i][j], j<N else 0 ----------
__global__ void transpose_f16(const float* __restrict__ src, _Float16* __restrict__ dst,
                              int K, int N, int Npad) {
    int id = blockIdx.x*blockDim.x + threadIdx.x;
    if (id >= K*Npad) return;
    int j = id / K, i = id % K;
    dst[id] = (j < N) ? (_Float16)src[(size_t)i*N + j] : (_Float16)0.f;
}

// ---------- Stage 2: conv-layer LIF recurrence -> spike train F[T][2496] (f16) ----------
__global__ void c1_rec(const float* __restrict__ Y, const float* __restrict__ Wc,
                       const float* __restrict__ bc, _Float16* __restrict__ F) {
    int id = blockIdx.x*blockDim.x + threadIdx.x;
    if (id >= NF) return;
    int n = id / F1, o = id % F1;
    float w[8];
    #pragma unroll
    for (int c8 = 0; c8 < 8; ++c8) w[c8] = Wc[c8*F1 + o];   // W_conv[k][c][o]
    float b = bc[o];
    const float* yb = Y + (size_t)n*8*T_STEPS;
    float m = 0.f, s = 0.f;
    for (int t = 0; t < T_STEPS; ++t) {
        float conv = b;
        #pragma unroll
        for (int c8 = 0; c8 < 8; ++c8) conv += w[c8] * yb[(size_t)c8*T_STEPS + t];
        m = m * DECAYC * (1.f - s) + conv;
        s = (m > THRESHC) ? 1.f : 0.f;
        F[(size_t)t*NF + id] = (_Float16)s;
    }
}

// ---------- WMMA GEMM: C[M,N] = A[M,K](f16) @ Bt[N,K]^T(f16); one wave per 16x16 tile ----
__global__ void gemm_wmma(const _Float16* __restrict__ A, const _Float16* __restrict__ Bt,
                          float* __restrict__ C, int M, int K, int N) {
    int tilesN = N >> 4;
    int tm = (int)blockIdx.x / tilesN;
    int tn = (int)blockIdx.x % tilesN;
    int lane = threadIdx.x & 31;
    int half = lane >> 4;     // 0: K-chunk base 0 / 1: base 8 (A and B)
    int lm   = lane & 15;
    const _Float16* arow = A  + (size_t)(tm*16 + lm)*K + half*8;
    const _Float16* brow = Bt + (size_t)(tn*16 + lm)*K + half*8;
    v8f acc = {};
    for (int kk = 0; kk < K; kk += 32) {
        union { v16h h; v4u q[2]; } a, b;
        a.q[0] = *(const v4u*)(arow + kk);        // K = c0..c0+7
        a.q[1] = *(const v4u*)(arow + kk + 16);   // K = c0+16..c0+23
        b.q[0] = *(const v4u*)(brow + kk);        // K = kb..kb+7  (column lm of B)
        b.q[1] = *(const v4u*)(brow + kk + 16);   // K = kb+16..kb+23
        acc = __builtin_amdgcn_wmma_f32_16x16x32_f16(
            false, a.h, false, b.h, (short)0, acc, false, false);
    }
    float* crow = C + (size_t)(tm*16 + half*8)*N + tn*16 + lm;
    #pragma unroll
    for (int r = 0; r < 8; ++r) crow[(size_t)r*N] = acc[r];   // row = tm*16 + 8*half + r
}

// ---------- dense-layer LIF recurrence; optional spike output / final average ----------
__global__ void h_rec(const float* __restrict__ G, const float* __restrict__ bias,
                      _Float16* __restrict__ Sout, float* __restrict__ out,
                      int Cn, int ldc) {
    int j = blockIdx.x*blockDim.x + threadIdx.x;
    if (j >= Cn) return;
    float b = bias[j];
    float m = 0.f, s = 0.f, acc = 0.f;
    for (int t = 0; t < T_STEPS; ++t) {
        m = m * DECAYC * (1.f - s) + G[(size_t)t*ldc + j] + b;
        s = (m > THRESHC) ? 1.f : 0.f;
        if (Sout) Sout[(size_t)t*Cn + j] = (_Float16)s;
        else      acc += s;
    }
    if (out) out[j] = acc / (float)T_STEPS;
}

extern "C" void kernel_launch(void* const* d_in, const int* in_sizes, int n_in,
                              void* d_out, int out_size, void* d_ws, size_t ws_size,
                              hipStream_t stream) {
    (void)in_sizes; (void)n_in; (void)out_size; (void)ws_size;
    const float* x  = (const float*)d_in[0];
    const int*   ei = (const int*)  d_in[1];
    const float* Wc = (const float*)d_in[2];
    const float* bc = (const float*)d_in[3];
    const float* W1 = (const float*)d_in[4];
    const float* b1 = (const float*)d_in[5];
    const float* W2 = (const float*)d_in[6];
    const float* b2 = (const float*)d_in[7];
    const float* W3 = (const float*)d_in[8];
    const float* b3 = (const float*)d_in[9];
    float* out = (float*)d_out;

    char* ws = (char*)d_ws;
    size_t off = 0;
    auto alloc = [&](size_t bytes) -> char* {
        char* p = ws + off;
        off = (off + bytes + 255) & ~(size_t)255;
        return p;
    };
    float*    Ahat = (float*)   alloc((size_t)NN*NN*4);
    float*    A2m  = (float*)   alloc((size_t)NN*NN*4);
    float*    A3m  = (float*)   alloc((size_t)NN*NN*4);
    float*    Y    = (float*)   alloc((size_t)NN*8*T_STEPS*4);      // 2.5 MB
    _Float16* W1t  = (_Float16*)alloc((size_t)H1*NF*2);             // 0.64 MB
    _Float16* W2t  = (_Float16*)alloc((size_t)H2*H1*2);
    _Float16* W3t  = (_Float16*)alloc((size_t)H3P*H2*2);
    _Float16* F    = (_Float16*)alloc((size_t)T_STEPS*NF*2);        // 10 MB spike train
    float*    G1   = (float*)   alloc((size_t)T_STEPS*H1*4);
    _Float16* S1   = (_Float16*)alloc((size_t)T_STEPS*H1*2);
    float*    G2   = (float*)   alloc((size_t)T_STEPS*H2*4);
    _Float16* S2   = (_Float16*)alloc((size_t)T_STEPS*H2*2);
    float*    G3   = (float*)   alloc((size_t)T_STEPS*H3P*4);

    // graph preprocessing
    build_ahat<<<1, 256, 0, stream>>>(ei, Ahat);
    matmul39<<<(NN*NN + 255)/256, 256, 0, stream>>>(Ahat, Ahat, A2m);
    matmul39<<<(NN*NN + 255)/256, 256, 0, stream>>>(A2m, Ahat, A3m);
    int ytot = NN*8*T_STEPS;
    build_ycat<<<(ytot + 255)/256, 256, 0, stream>>>(x, Ahat, A2m, A3m, Y);

    // weight transpose + f16 convert
    transpose_f16<<<(NF*H1  + 255)/256, 256, 0, stream>>>(W1, W1t, NF, H1, H1);
    transpose_f16<<<(H1*H2  + 255)/256, 256, 0, stream>>>(W2, W2t, H1, H2, H2);
    transpose_f16<<<(H2*H3P + 255)/256, 256, 0, stream>>>(W3, W3t, H2, H3, H3P);

    // conv-layer recurrence -> F; then alternate WMMA GEMM / recurrence
    c1_rec<<<(NF + 255)/256, 256, 0, stream>>>(Y, Wc, bc, F);
    gemm_wmma<<<(T_STEPS/16)*(H1/16),  32, 0, stream>>>(F,  W1t, G1, T_STEPS, NF, H1);
    h_rec<<<1, H1, 0, stream>>>(G1, b1, S1, nullptr, H1, H1);
    gemm_wmma<<<(T_STEPS/16)*(H2/16),  32, 0, stream>>>(S1, W2t, G2, T_STEPS, H1, H2);
    h_rec<<<1, H2, 0, stream>>>(G2, b2, S2, nullptr, H2, H2);
    gemm_wmma<<<(T_STEPS/16)*(H3P/16), 32, 0, stream>>>(S2, W3t, G3, T_STEPS, H2, H3P);
    h_rec<<<1, 64, 0, stream>>>(G3, b3, nullptr, out, H3, H3P);
}
